// SessionAwareFusion_7490422964560
// MI455X (gfx1250) — matrix operand order repaired and checked by reading, not compiled, not run on hardware
//
#include <hip/hip_runtime.h>
#include <hip/hip_bf16.h>
#include <math.h>

// ---------------------------------------------------------------------------
// SessionAwareFusion forward, fused for gfx1250 (MI455X, wave32, WMMA + TDM).
//   B=32768, D=256, N=4, OUT=512, H=4, DH=64, CAT=1024
// One workgroup = 16 batch rows, 128 threads (4 waves). All GEMMs use
// v_wmma_f32_16x16x32_f16 (f32 accum), 2 N-tiles per wave with a software-
// pipelined K loop. Input tiles staged to LDS via TENSOR_LOAD_TO_LDS (TDM).
// ---------------------------------------------------------------------------

typedef __attribute__((ext_vector_type(16))) _Float16 v16h;
typedef __attribute__((ext_vector_type(8)))  float    v8f;
typedef __attribute__((ext_vector_type(4)))  unsigned int v4u;
typedef __attribute__((ext_vector_type(8)))  int      v8i;
typedef __attribute__((ext_vector_type(4)))  int      v4i;

#define BSZ   32768
#define DM    256
#define NTOK  4
#define OUTM  512
#define HH    4
#define DHD   64
#define CATW  1024
#define ROWS  16
#define NTHREADS 128
#define NWAVES   4

// ---- LDS layout (byte offsets), total 122880 B -----------------------------
// XH0   @      0 : 16x256  f16 (scaled w0 = query)                 8192 B
// CAT   @   8192 : 16x1024 f16 (entry | x1 | x2 | x3)             32768 B
// QH    @  40960 : 16x256  f16                                     8192 B
// KH    @  49152 : 4 x 16x256 f16  (TDM staging first, then HB)   32768 B
// VH    @  81920 : 4 x 16x256 f16  (TDM staging / ENTRYF / FUSEDF)32768 B
// CTXH  @ 114688 : 16x256  f16     (reused as AH 16x128)           8192 B
#define LDS_BYTES 122880
#define STAGE_OFF 49152u     // 4 x 16KB f32 staging tiles (over KH+VH)

// A fragment: 16x32 f16, ISA 7.12.2 layout.
__device__ __forceinline__ v16h load_frag_a(const _Float16* base, int ldh, int lane) {
    int r  = lane & 15;
    int kb = (lane >> 4) << 3;          // 0 or 8
    const _Float16* p = base + r * ldh + kb;
    union { v16h v; float4 q[2]; } u;
    u.q[0] = *(const float4*)(p);       // K = kb .. kb+7
    u.q[1] = *(const float4*)(p + 16);  // K = kb+16 .. kb+23
    return u.v;
}

// B fragment: 32x16 f16 (K x N); B column n is W row n (we compute X@W^T).
__device__ __forceinline__ v16h load_frag_b(const _Float16* wrow_base, int ldh, int lane) {
    int n  = lane & 15;
    int kb = (lane >> 4) << 3;          // 0 or 8
    const _Float16* p = wrow_base + n * ldh + kb;
    union { v16h v; float4 q[2]; } u;
    u.q[0] = *(const float4*)(p);
    u.q[1] = *(const float4*)(p + 16);
    return u.v;
}

__device__ __forceinline__ v8f wmma_f16(v8f acc, v16h a, v16h b) {
    return __builtin_amdgcn_wmma_f32_16x16x32_f16(
        false, a, false, b, (short)0, acc, false, false);
}

// Software-pipelined dual-N-tile GEMM: two 16x16 output tiles sharing one A
// fragment per K step. Next iteration's fragments are loaded before the
// current WMMAs so waits cover only the older loads.
__device__ __forceinline__ void gemm_k2(const _Float16* A, int lda,
                                        const _Float16* W0, const _Float16* W1,
                                        int ldw, int K, int lane,
                                        v8f& acc0, v8f& acc1) {
    v16h a  = load_frag_a(A, lda, lane);
    v16h b0 = load_frag_b(W0, ldw, lane);
    v16h b1 = load_frag_b(W1, ldw, lane);
    #pragma unroll 4
    for (int kk = 32; kk < K; kk += 32) {
        v16h an  = load_frag_a(A + kk, lda, lane);
        v16h b0n = load_frag_b(W0 + kk, ldw, lane);
        v16h b1n = load_frag_b(W1 + kk, ldw, lane);
        acc0 = wmma_f16(acc0, a, b0);
        acc1 = wmma_f16(acc1, a, b1);
        a = an; b0 = b0n; b1 = b1n;
    }
    acc0 = wmma_f16(acc0, a, b0);
    acc1 = wmma_f16(acc1, a, b1);
}

__device__ __forceinline__ float gelu_exact(float x) {
    return 0.5f * x * (1.0f + erff(x * 0.70710678118654752f));
}

// TDM: load one 16x256 f32 tile (row-major, stride 256) into LDS.
// D# per CDNA5 ISA ch.8: group0 = {count=1, lds_addr, gaddr_lo, gaddr_hi|type2}
// group1: data_size=4B(2), tensor_dim0=256, tensor_dim1=32768, tile=256x16,
// tensor_dim0_stride=256.
__device__ __forceinline__ void tdm_load_tile(const float* gsrc, unsigned lds_off) {
    unsigned long long ga = (unsigned long long)(const void*)gsrc;
    v4u g0 = { 1u, lds_off, (unsigned)(ga & 0xffffffffu),
               (unsigned)(((ga >> 32) & 0x01ffffffu) | 0x80000000u) };
    v8i g1 = { 0x00020000, 0x01000000, (int)0x80000000, 0x01000000,
               16, 256, 0, 0 };
    v4i g2 = { 0, 0, 0, 0 };
    v4i g3 = { 0, 0, 0, 0 };
#if defined(__clang_major__) && __clang_major__ >= 23
    v8i g4 = { 0, 0, 0, 0, 0, 0, 0, 0 };
    __builtin_amdgcn_tensor_load_to_lds(g0, g1, g2, g3, g4, 0);
#else
    __builtin_amdgcn_tensor_load_to_lds(g0, g1, g2, g3, 0);
#endif
}

// ---------------------------------------------------------------------------
__global__ void __launch_bounds__(NTHREADS)
fused_forward(const float* __restrict__ w0, const float* __restrict__ w1,
              const float* __restrict__ w2, const float* __restrict__ w3,
              const float* __restrict__ regime, const float* __restrict__ tfw,
              const _Float16* __restrict__ wqkvh, const float* __restrict__ in_proj_b,
              const _Float16* __restrict__ outph, const float* __restrict__ out_proj_b,
              const float* __restrict__ ln1_g, const float* __restrict__ ln1_b,
              const float* __restrict__ sg_w, const float* __restrict__ sg_b,
              const float* __restrict__ sb_w, const float* __restrict__ sb_b,
              const _Float16* __restrict__ f1h, const float* __restrict__ f1_b,
              const _Float16* __restrict__ f2h, const float* __restrict__ f2_b,
              const float* __restrict__ ln2_g, const float* __restrict__ ln2_b,
              const _Float16* __restrict__ a1h, const float* __restrict__ a1_b,
              const float* __restrict__ a2_w, const float* __restrict__ a2_b,
              float* __restrict__ out_fused, float* __restrict__ out_align)
{
    extern __shared__ char smem[];
    _Float16* XH0    = (_Float16*)(smem);
    _Float16* CAT    = (_Float16*)(smem + 8192);
    _Float16* QH     = (_Float16*)(smem + 40960);
    _Float16* KH     = (_Float16*)(smem + 49152);
    _Float16* VH     = (_Float16*)(smem + 81920);
    _Float16* CTXH   = (_Float16*)(smem + 114688);
    const float* SF32= (const float*)(smem + STAGE_OFF); // 4 x 16x256 f32 staging
    float*    ENTRYF = (float*)   (smem + 81920);   // over VH  (16x256 f32)
    _Float16* HB     = (_Float16*)(smem + 49152);   // over KH  (16x1024)
    float*    FUSEDF = (float*)   (smem + 81920);   // over VH  (16x512 f32)
    _Float16* AH     = (_Float16*)(smem + 114688);  // over CTXH (16x128)

    const int tid  = threadIdx.x;
    const int lane = tid & 31;
    const int wave = tid >> 5;
    const int b0   = blockIdx.x * ROWS;

    // ---- Phase 0: TDM async-stage the four input tiles into LDS --------
    if (wave == 0) {
        tdm_load_tile(w0 + (size_t)b0 * DM, STAGE_OFF);
        tdm_load_tile(w1 + (size_t)b0 * DM, STAGE_OFF + 16384u);
        tdm_load_tile(w2 + (size_t)b0 * DM, STAGE_OFF + 32768u);
        tdm_load_tile(w3 + (size_t)b0 * DM, STAGE_OFF + 49152u);
        __builtin_amdgcn_s_wait_tensorcnt(0);
    }

    // softmax(tf_weights) — 4 scalars, recomputed per thread (L2 hit)
    float t0 = tfw[0], t1 = tfw[1], t2 = tfw[2], t3 = tfw[3];
    float tm = fmaxf(fmaxf(t0, t1), fmaxf(t2, t3));
    float e0 = __expf(t0 - tm), e1 = __expf(t1 - tm),
          e2 = __expf(t2 - tm), e3 = __expf(t3 - tm);
    float einv = 1.0f / (e0 + e1 + e2 + e3);
    const float s0 = e0 * einv, s1 = e1 * einv, s2 = e2 * einv, s3 = e3 * einv;
    __syncthreads();

    // ---- Phase 1: scale + cvt staged tiles to f16 ----------------------
    for (int idx = tid; idx < ROWS * DM; idx += NTHREADS) {
        int m = idx >> 8, k = idx & 255;
        int o = m * DM + k;
        XH0[o]                   = (_Float16)(SF32[o]          * s0);
        CAT[m * CATW + 256 + k]  = (_Float16)(SF32[4096  + o - m * DM + m * DM] * s1);
        CAT[m * CATW + 512 + k]  = (_Float16)(SF32[8192  + o]  * s2);
        CAT[m * CATW + 768 + k]  = (_Float16)(SF32[12288 + o]  * s3);
    }
    __syncthreads();

    // ---- Phase 2: QKV projections (WMMA, paired N-tiles) ---------------
    // pair-jobs: 0..7 Q, 8..39 K (4 tok x 8 pairs), 40..71 V
    for (int pj = wave; pj < 72; pj += NWAVES) {
        const _Float16* A; int lda; _Float16* Dst; int brow0;
        if (pj < 8) {
            A = XH0; lda = DM; Dst = QH; brow0 = pj * 32;
        } else if (pj < 40) {
            int t = pj - 8; int j = t >> 3; int np = t & 7;
            A   = (j == 0) ? XH0 : (CAT + j * 256);
            lda = (j == 0) ? DM : CATW;
            Dst = KH + j * ROWS * DM; brow0 = 256 + np * 32;
        } else {
            int t = pj - 40; int j = t >> 3; int np = t & 7;
            A   = (j == 0) ? XH0 : (CAT + j * 256);
            lda = (j == 0) ? DM : CATW;
            Dst = VH + j * ROWS * DM; brow0 = 512 + np * 32;
        }
        const _Float16* W0 = wqkvh + (size_t)brow0 * DM;
        v8f acc0 = {}, acc1 = {};
        gemm_k2(A, lda, W0, W0 + 16 * DM, DM, DM, lane, acc0, acc1);
        int nl = lane & 15, mb = (lane >> 4) * 8;
        int ng = (brow0 & 255) + nl;                  // column within 0..255
        float bi0 = in_proj_b[brow0 + nl];
        float bi1 = in_proj_b[brow0 + 16 + nl];
        #pragma unroll
        for (int i = 0; i < 8; i++) {
            Dst[(mb + i) * DM + ng]      = (_Float16)(acc0[i] + bi0);
            Dst[(mb + i) * DM + ng + 16] = (_Float16)(acc1[i] + bi1);
        }
    }
    __syncthreads();

    // ---- Phase 3: tiny attention (N=4 tokens) per (row, head) ----------
    if (tid < ROWS * HH) {
        int m = tid >> 2, h = tid & 3;
        const _Float16* q = QH + m * DM + h * DHD;
        float sc[NTOK];
        #pragma unroll
        for (int j = 0; j < NTOK; j++) {
            const _Float16* kp = KH + j * ROWS * DM + m * DM + h * DHD;
            float d = 0.f;
            for (int x = 0; x < DHD; x++) d += (float)q[x] * (float)kp[x];
            sc[j] = d * 0.125f;                     // 1/sqrt(64)
        }
        float mx = fmaxf(fmaxf(sc[0], sc[1]), fmaxf(sc[2], sc[3]));
        float ex[NTOK], es = 0.f;
        #pragma unroll
        for (int j = 0; j < NTOK; j++) { ex[j] = __expf(sc[j] - mx); es += ex[j]; }
        float esi = 1.0f / es;
        _Float16* c = CTXH + m * DM + h * DHD;
        for (int x = 0; x < DHD; x++) {
            float v = 0.f;
            #pragma unroll
            for (int j = 0; j < NTOK; j++)
                v += ex[j] * esi * (float)VH[j * ROWS * DM + m * DM + h * DHD + x];
            c[x] = (_Float16)v;
        }
    }
    __syncthreads();

    // ---- Phase 4: out_proj (WMMA) + residual, f32 into ENTRYF ----------
    for (int p = wave; p < 8; p += NWAVES) {
        int nt0 = p * 2;
        const _Float16* W0 = outph + (size_t)nt0 * 16 * DM;
        v8f acc0 = {}, acc1 = {};
        gemm_k2(CTXH, DM, W0, W0 + 16 * DM, DM, DM, lane, acc0, acc1);
        int nl = lane & 15, mb = (lane >> 4) * 8;
        int ng = nt0 * 16 + nl;
        float bi0 = out_proj_b[ng], bi1 = out_proj_b[ng + 16];
        #pragma unroll
        for (int i = 0; i < 8; i++) {
            int m = mb + i;
            ENTRYF[m * DM + ng]      = acc0[i] + bi0 + (float)XH0[m * DM + ng];
            ENTRYF[m * DM + ng + 16] = acc1[i] + bi1 + (float)XH0[m * DM + ng + 16];
        }
    }
    __syncthreads();

    // ---- Phase 5: LayerNorm1 + FiLM, write entry into CAT[:,0:256] -----
    {
        int g = lane >> 3, sub = lane & 7, m = wave * 4 + g;
        int b = b0 + m;
        float sum = 0.f, ss = 0.f;
        for (int c = sub; c < DM; c += 8) {
            float x = ENTRYF[m * DM + c]; sum += x; ss += x * x;
        }
        #pragma unroll
        for (int o = 1; o < 8; o <<= 1) {
            sum += __shfl_xor(sum, o, 32);
            ss  += __shfl_xor(ss,  o, 32);
        }
        float mu  = sum * (1.0f / DM);
        float var = ss * (1.0f / DM) - mu * mu;
        float rs  = rsqrtf(var + 1e-5f);
        float r0 = regime[b * 4 + 0], r1 = regime[b * 4 + 1],
              r2 = regime[b * 4 + 2], r3 = regime[b * 4 + 3];
        for (int c = sub; c < DM; c += 8) {
            float x  = (ENTRYF[m * DM + c] - mu) * rs * ln1_g[c] + ln1_b[c];
            float ga = r0 * sg_w[c * 4 + 0] + r1 * sg_w[c * 4 + 1] +
                       r2 * sg_w[c * 4 + 2] + r3 * sg_w[c * 4 + 3] + sg_b[c];
            float be = r0 * sb_w[c * 4 + 0] + r1 * sb_w[c * 4 + 1] +
                       r2 * sb_w[c * 4 + 2] + r3 * sb_w[c * 4 + 3] + sb_b[c];
            CAT[m * CATW + c] = (_Float16)(x * (1.0f + ga) + be);
        }
    }
    __syncthreads();

    // ---- Phase 6: h = gelu(cat @ f1^T + b1), 16x1024 (WMMA) ------------
    for (int p = wave; p < 32; p += NWAVES) {
        int nt0 = p * 2;
        const _Float16* W0 = f1h + (size_t)nt0 * 16 * CATW;
        __builtin_prefetch(W0, 0, 0);                 // global_prefetch_b8
        __builtin_prefetch(W0 + 16 * CATW, 0, 0);
        v8f acc0 = {}, acc1 = {};
        gemm_k2(CAT, CATW, W0, W0 + 16 * CATW, CATW, CATW, lane, acc0, acc1);
        int nl = lane & 15, mb = (lane >> 4) * 8;
        int ng = nt0 * 16 + nl;
        float bi0 = f1_b[ng], bi1 = f1_b[ng + 16];
        #pragma unroll
        for (int i = 0; i < 8; i++) {
            HB[(mb + i) * CATW + ng]      = (_Float16)gelu_exact(acc0[i] + bi0);
            HB[(mb + i) * CATW + ng + 16] = (_Float16)gelu_exact(acc1[i] + bi1);
        }
    }
    __syncthreads();

    // ---- Phase 7: fused_pre = h @ f2^T + b2, 16x512 f32 (WMMA) ---------
    for (int p = wave; p < 16; p += NWAVES) {
        int nt0 = p * 2;
        const _Float16* W0 = f2h + (size_t)nt0 * 16 * CATW;
        v8f acc0 = {}, acc1 = {};
        gemm_k2(HB, CATW, W0, W0 + 16 * CATW, CATW, CATW, lane, acc0, acc1);
        int nl = lane & 15, mb = (lane >> 4) * 8;
        int ng = nt0 * 16 + nl;
        float bi0 = f2_b[ng], bi1 = f2_b[ng + 16];
        #pragma unroll
        for (int i = 0; i < 8; i++) {
            FUSEDF[(mb + i) * OUTM + ng]      = acc0[i] + bi0;
            FUSEDF[(mb + i) * OUTM + ng + 16] = acc1[i] + bi1;
        }
    }
    __syncthreads();

    // ---- Phase 8: LayerNorm2 -> d_out (fused region, f32) --------------
    {
        int g = lane >> 3, sub = lane & 7, m = wave * 4 + g;
        float sum = 0.f, ss = 0.f;
        for (int c = sub; c < OUTM; c += 8) {
            float x = FUSEDF[m * OUTM + c]; sum += x; ss += x * x;
        }
        #pragma unroll
        for (int o = 1; o < 8; o <<= 1) {
            sum += __shfl_xor(sum, o, 32);
            ss  += __shfl_xor(ss,  o, 32);
        }
        float mu  = sum * (1.0f / OUTM);
        float var = ss * (1.0f / OUTM) - mu * mu;
        float rs  = rsqrtf(var + 1e-5f);
        for (int c = sub; c < OUTM; c += 8) {
            float x = (FUSEDF[m * OUTM + c] - mu) * rs * ln2_g[c] + ln2_b[c];
            out_fused[(size_t)(b0 + m) * OUTM + c] = x;
        }
    }

    // ---- Phase 9: a = gelu(cat @ a1^T + b), 16x128 (WMMA) --------------
    for (int p = wave; p < 4; p += NWAVES) {
        int nt0 = p * 2;
        const _Float16* W0 = a1h + (size_t)nt0 * 16 * CATW;
        v8f acc0 = {}, acc1 = {};
        gemm_k2(CAT, CATW, W0, W0 + 16 * CATW, CATW, CATW, lane, acc0, acc1);
        int nl = lane & 15, mb = (lane >> 4) * 8;
        int ng = nt0 * 16 + nl;
        float bi0 = a1_b[ng], bi1 = a1_b[ng + 16];
        #pragma unroll
        for (int i = 0; i < 8; i++) {
            AH[(mb + i) * 128 + ng]      = (_Float16)gelu_exact(acc0[i] + bi0);
            AH[(mb + i) * 128 + ng + 16] = (_Float16)gelu_exact(acc1[i] + bi1);
        }
    }
    __syncthreads();

    // ---- Phase 10: alignment = sigmoid(a @ a2^T + b) -------------------
    if (tid < ROWS) {
        int m = tid;
        float acc = a2_b[0];
        for (int i = 0; i < 128; i++)
            acc += (float)AH[m * 128 + i] * a2_w[i];
        out_align[b0 + m] = 1.0f / (1.0f + __expf(-acc));
    }
}

// ---------------------------------------------------------------------------
// One-time weight conversion fp32 -> f16 into workspace (3.93 MB).
// ---------------------------------------------------------------------------
#define SEG0 196608u
#define SEG1 (SEG0 + 65536u)
#define SEG2 (SEG1 + 1048576u)
#define SEG3 (SEG2 + 524288u)
#define SEG4 (SEG3 + 131072u)

__global__ void __launch_bounds__(256)
cvt_weights(const float* __restrict__ in_proj_w, const float* __restrict__ out_proj_w,
            const float* __restrict__ f1_w, const float* __restrict__ f2_w,
            const float* __restrict__ a1_w, _Float16* __restrict__ dst)
{
    unsigned i = blockIdx.x * 256u + threadIdx.x;
    if (i >= SEG4) return;
    float v;
    if      (i < SEG0) v = in_proj_w[i];
    else if (i < SEG1) v = out_proj_w[i - SEG0];
    else if (i < SEG2) v = f1_w[i - SEG1];
    else if (i < SEG3) v = f2_w[i - SEG2];
    else               v = a1_w[i - SEG3];
    dst[i] = (_Float16)v;
}

// ---------------------------------------------------------------------------
extern "C" void kernel_launch(void* const* d_in, const int* in_sizes, int n_in,
                              void* d_out, int out_size, void* d_ws, size_t ws_size,
                              hipStream_t stream)
{
    (void)in_sizes; (void)n_in; (void)out_size; (void)ws_size;
    const float* w0         = (const float*)d_in[0];
    const float* w1         = (const float*)d_in[1];
    const float* w2         = (const float*)d_in[2];
    const float* w3         = (const float*)d_in[3];
    const float* regime     = (const float*)d_in[4];
    const float* tf_weights = (const float*)d_in[5];
    const float* in_proj_w  = (const float*)d_in[6];
    const float* in_proj_b  = (const float*)d_in[7];
    const float* out_proj_w = (const float*)d_in[8];
    const float* out_proj_b = (const float*)d_in[9];
    const float* ln1_g      = (const float*)d_in[10];
    const float* ln1_b      = (const float*)d_in[11];
    const float* sg_w       = (const float*)d_in[12];
    const float* sg_b       = (const float*)d_in[13];
    const float* sb_w       = (const float*)d_in[14];
    const float* sb_b       = (const float*)d_in[15];
    const float* f1_w       = (const float*)d_in[16];
    const float* f1_b       = (const float*)d_in[17];
    const float* f2_w       = (const float*)d_in[18];
    const float* f2_b       = (const float*)d_in[19];
    const float* ln2_g      = (const float*)d_in[20];
    const float* ln2_b      = (const float*)d_in[21];
    const float* a1_w       = (const float*)d_in[22];
    const float* a1_b       = (const float*)d_in[23];
    const float* a2_w       = (const float*)d_in[24];
    const float* a2_b       = (const float*)d_in[25];

    _Float16* wh = (_Float16*)d_ws;
    const _Float16* wqkvh = wh;
    const _Float16* outph = wh + SEG0;
    const _Float16* f1h   = wh + SEG1;
    const _Float16* f2h   = wh + SEG2;
    const _Float16* a1h   = wh + SEG3;

    float* out_fused = (float*)d_out;
    float* out_align = out_fused + (size_t)BSZ * OUTM;

    cvt_weights<<<(SEG4 + 255u) / 256u, 256, 0, stream>>>(
        in_proj_w, out_proj_w, f1_w, f2_w, a1_w, wh);

    fused_forward<<<BSZ / ROWS, NTHREADS, LDS_BYTES, stream>>>(
        w0, w1, w2, w3, regime, tf_weights,
        wqkvh, in_proj_b, outph, out_proj_b,
        ln1_g, ln1_b, sg_w, sg_b, sb_w, sb_b,
        f1h, f1_b, f2h, f2_b, ln2_g, ln2_b,
        a1h, a1_b, a2_w, a2_b,
        out_fused, out_align);
}
